// TimeAwareRNN_40621800685589
// MI455X (gfx1250) — compile-verified, matrix operand order checked by reading, not run on hardware
//
#include <hip/hip_runtime.h>
#include <hip/hip_bf16.h>
#include <math.h>

typedef __attribute__((ext_vector_type(16))) __bf16 v16bf;
typedef __attribute__((ext_vector_type(8)))  float  v8f;

#define BB 512
#define TT 200
#define DD 128
#define HH 256
#define KCAT 384            // D + H : fused [x_t | h] A operand
#define NGATE 1024          // 4*H gate columns

#define LOG2E 1.4426950408889634f

// Branch-free activations on the hardware transcendental pipe.
__device__ __forceinline__ float fast_exp(float x) {
    return __builtin_amdgcn_exp2f(x * LOG2E);
}
__device__ __forceinline__ float fast_sigmoid(float x) {
    // rcp(1 + 2^(-x*log2e)); saturates to 0/1 at the extremes.
    return __builtin_amdgcn_rcpf(1.0f + __builtin_amdgcn_exp2f(-x * LOG2E));
}
__device__ __forceinline__ float fast_tanh(float x) {
    // 1 - 2/(e^{2x}+1); saturates to +/-1 without NaN at the extremes.
    float e = __builtin_amdgcn_exp2f(x * (2.0f * LOG2E));
    return 1.0f - 2.0f * __builtin_amdgcn_rcpf(e + 1.0f);
}

// ---------------------------------------------------------------------------
// Kernel 1: convert + swizzle weights to bf16 WMMA B-operand blocks.
// Block layout: (kt,nt) 32x16 bf16 block = 512 elems; within block lane l
// holds 16 contiguous bf16: lane = 16*(krow>=16) + ncol, idx = krow%16.
// Each WMMA B-fragment load is then lane*32B contiguous -> coalesced b128s.
// ---------------------------------------------------------------------------
__global__ __launch_bounds__(256)
void prep_kernel(const float* __restrict__ Wxf, const float* __restrict__ Wxi,
                 const float* __restrict__ Wxo, const float* __restrict__ Wxg,
                 const float* __restrict__ Whf, const float* __restrict__ Whi,
                 const float* __restrict__ Who, const float* __restrict__ Whg,
                 const float* __restrict__ Wcd,
                 const float* __restrict__ bxf, const float* __restrict__ bxi,
                 const float* __restrict__ bxo, const float* __restrict__ bxg,
                 __bf16* __restrict__ wcat, __bf16* __restrict__ wcdb,
                 float* __restrict__ bx)
{
    const int NW = KCAT * NGATE;      // 393216
    const int NC = HH * HH;           // 65536
    int e = blockIdx.x * 256 + threadIdx.x;
    if (e < NW) {
        int k = e >> 10, n = e & 1023;
        int g = n >> 8, col = n & 255;
        const float* src;
        if (k < DD) src = (g == 0 ? Wxf : g == 1 ? Wxi : g == 2 ? Wxo : Wxg) + k * HH;
        else        src = (g == 0 ? Whf : g == 1 ? Whi : g == 2 ? Who : Whg) + (k - DD) * HH;
        float v = src[col];
        int kt = k >> 5, kr = k & 31, nt = n >> 4, c = n & 15;
        int lane = ((kr >= 16) ? 16 : 0) + c;
        wcat[((size_t)(kt * 64 + nt)) * 512 + lane * 16 + (kr & 15)] = (__bf16)v;
    } else if (e < NW + NC) {
        int e2 = e - NW;
        int k = e2 >> 8, n = e2 & 255;
        float v = Wcd[k * HH + n];
        int kt = k >> 5, kr = k & 31, nt = n >> 4, c = n & 15;
        int lane = ((kr >= 16) ? 16 : 0) + c;
        wcdb[((size_t)(kt * 16 + nt)) * 512 + lane * 16 + (kr & 15)] = (__bf16)v;
    } else if (e < NW + NC + NGATE) {
        int n = e - NW - NC;
        const float* b = (n < 256 ? bxf : n < 512 ? bxi : n < 768 ? bxo : bxg);
        bx[n] = b[n & 255];
    }
}

// ---------------------------------------------------------------------------
// Kernel 2: recurrent scan. One workgroup per 16 batch rows (WMMA M-tile),
// 8 waves; wave w owns H-columns [32w, 32w+32). Per step, per wave:
//   2x8  WMMA  : c_short = tanh(c @ Wcd + bcd)          (K=256)
//   8x12 WMMA  : gates   = [x_t|h] @ Wcat + bx          (K=384, A reused)
// First K iteration of each GEMM is peeled so the accumulators start from
// the WMMA inline-constant 0 (no per-step zero-fill movs). K-loops rolled
// (#pragma unroll 1) to keep the live fragment set small (no spills).
// ---------------------------------------------------------------------------
__global__ __launch_bounds__(256)
void scan_kernel(const float* __restrict__ inputs,
                 const __bf16* __restrict__ wcat,
                 const __bf16* __restrict__ wcdb,
                 const float* __restrict__ bx,
                 const float* __restrict__ bcd,
                 float* __restrict__ hs)
{
    __shared__ __attribute__((aligned(32))) __bf16 a_buf[16][KCAT]; // [x_t | h]
    __shared__ __attribute__((aligned(32))) __bf16 c_bf[16][HH];    // c as bf16 A-operand
    __shared__ float c_f[16][HH];
    __shared__ float rdt_s[16];          // 1/(dt_t + 1)

    const int tid   = threadIdx.x;
    const int wave  = tid >> 5;
    const int lane  = tid & 31;
    const int bbase = blockIdx.x * 16;

    const int mrow  = lane & 15;        // A-operand row this lane reads
    const int khalf = lane >> 4;        // A-operand K-half
    const int ncol  = lane & 15;        // C-layout: N within tile
    const int mbase = (lane >> 4) * 8;  // C-layout: M = r + mbase

    // init h = 0, c = 0
    for (int i = tid; i < 16 * HH; i += 256) {
        int m = i >> 8, n = i & 255;
        c_f[m][n] = 0.0f;
        c_bf[m][n] = (__bf16)0.0f;
        a_buf[m][DD + n] = (__bf16)0.0f;
    }
    __syncthreads();

    const v8f vzero = {0.f, 0.f, 0.f, 0.f, 0.f, 0.f, 0.f, 0.f};

    for (int t = 0; t < TT; ++t) {
        // ---- stage x_t (16x128) into A buffer + 1/(dt+1) ----
        {
            int m  = tid >> 4;
            int d0 = (tid & 15) * 8;
            const float* src = inputs + (((size_t)(bbase + m)) * TT + t) * (DD + 1) + d0;
#pragma unroll
            for (int q = 0; q < 8; ++q) a_buf[m][d0 + q] = (__bf16)src[q];
            if (tid < 16) {
                float dt = inputs[(((size_t)(bbase + tid)) * TT + t) * (DD + 1) + DD];
                rdt_s[tid] = __builtin_amdgcn_rcpf(dt + 1.0f);
            }
        }
        __syncthreads();

        // ---- c_short -> c_adj (2 tiles per wave, kept in registers) ----
        float adj[2][8];
        {
            const int nt0 = 2 * wave, nt1 = 2 * wave + 1;
            v8f acc0, acc1;
            {   // peeled kt = 0: C = inline 0
                v16bf a  = *(const v16bf*)&c_bf[mrow][khalf * 16];
                v16bf b0 = *(const v16bf*)(wcdb + ((size_t)nt0) * 512 + lane * 16);
                v16bf b1 = *(const v16bf*)(wcdb + ((size_t)nt1) * 512 + lane * 16);
                acc0 = __builtin_amdgcn_wmma_f32_16x16x32_bf16(
                         false, a, false, b0, (short)0, vzero, false, false);
                acc1 = __builtin_amdgcn_wmma_f32_16x16x32_bf16(
                         false, a, false, b1, (short)0, vzero, false, false);
            }
#pragma unroll 1
            for (int kt = 1; kt < 8; ++kt) {
                v16bf a  = *(const v16bf*)&c_bf[mrow][kt * 32 + khalf * 16];
                v16bf b0 = *(const v16bf*)(wcdb + ((size_t)(kt * 16 + nt0)) * 512 + lane * 16);
                v16bf b1 = *(const v16bf*)(wcdb + ((size_t)(kt * 16 + nt1)) * 512 + lane * 16);
                acc0 = __builtin_amdgcn_wmma_f32_16x16x32_bf16(
                         false, a, false, b0, (short)0, acc0, false, false);
                acc1 = __builtin_amdgcn_wmma_f32_16x16x32_bf16(
                         false, a, false, b1, (short)0, acc1, false, false);
            }
#pragma unroll
            for (int j = 0; j < 2; ++j) {
                int n = (2 * wave + j) * 16 + ncol;
                float bc = bcd[n];
#pragma unroll
                for (int r = 0; r < 8; ++r) {
                    int m = r + mbase;
                    float a_ = (j == 0) ? acc0[r] : acc1[r];
                    float cs   = fast_tanh(a_ + bc);
                    float cold = c_f[m][n];
                    adj[j][r]  = cold - cs + cs * rdt_s[m];
                }
            }
        }

        // ---- gates = [x_t|h] @ Wcat : 8 tiles (4 gates x 2), K=384 ----
        v8f acc[4][2];
        {   // peeled kt = 0: C = inline 0
            v16bf a = *(const v16bf*)&a_buf[mrow][khalf * 16];
#pragma unroll
            for (int g = 0; g < 4; ++g) {
#pragma unroll
                for (int j = 0; j < 2; ++j) {
                    int nt = g * 16 + 2 * wave + j;
                    v16bf b = *(const v16bf*)(wcat + ((size_t)nt) * 512 + lane * 16);
                    acc[g][j] = __builtin_amdgcn_wmma_f32_16x16x32_bf16(
                                    false, a, false, b, (short)0, vzero, false, false);
                }
            }
        }
#pragma unroll 1
        for (int kt = 1; kt < 12; ++kt) {
            v16bf a = *(const v16bf*)&a_buf[mrow][kt * 32 + khalf * 16]; // reused 8x
#pragma unroll
            for (int g = 0; g < 4; ++g) {
#pragma unroll
                for (int j = 0; j < 2; ++j) {
                    int nt = g * 16 + 2 * wave + j;      // N0 = 256g + 32w + 16j
                    v16bf b = *(const v16bf*)(wcat + ((size_t)(kt * 64 + nt)) * 512 + lane * 16);
                    acc[g][j] = __builtin_amdgcn_wmma_f32_16x16x32_bf16(
                                    false, a, false, b, (short)0, acc[g][j], false, false);
                }
            }
        }

        __syncthreads();   // all waves done reading a_buf / c_f before update

        // ---- LSTM update in registers; write back h, c ----
#pragma unroll
        for (int j = 0; j < 2; ++j) {
            int n = (2 * wave + j) * 16 + ncol;          // column within H
            float bf_ = bx[n], bi_ = bx[256 + n], bo_ = bx[512 + n], bg_ = bx[768 + n];
#pragma unroll
            for (int r = 0; r < 8; ++r) {
                int m = r + mbase;
                float f  = fast_sigmoid(acc[0][j][r] + bf_);
                float ii = fast_sigmoid(acc[1][j][r] + bi_);
                float o  = fast_sigmoid(acc[2][j][r] + bo_);
                float g  = fast_tanh(acc[3][j][r] + bg_);
                float cn = f * adj[j][r] + ii * g;
                float hn = o * fast_tanh(cn);
                c_f[m][n]        = cn;
                c_bf[m][n]       = (__bf16)cn;
                a_buf[m][DD + n] = (__bf16)hn;
                hs[(((size_t)(bbase + m)) * TT + t) * HH + n] = hn;
            }
        }
        __syncthreads();
    }
}

// ---------------------------------------------------------------------------
// Kernel 3: attention pooling. One workgroup per batch row.
// score_t = hs_t . Wv1[0:256] + (h_last . Wv1[256:512] + bv1)  (const over t)
// out = sum_t softmax(score)_t * hs_t
// ---------------------------------------------------------------------------
__global__ __launch_bounds__(256)
void pool_kernel(const float* __restrict__ hs, const float* __restrict__ Wv1,
                 const float* __restrict__ bv1, float* __restrict__ out)
{
    __shared__ float sc[TT];
    __shared__ float red[8];
    __shared__ float w1[HH];
    __shared__ float s_c, s_max, s_sum;

    const int tid = threadIdx.x, wave = tid >> 5, lane = tid & 31;
    const int b = blockIdx.x;
    const float* hb = hs + (size_t)b * TT * HH;

    w1[tid] = Wv1[tid];
    float p = hb[(TT - 1) * HH + tid] * Wv1[HH + tid];
#pragma unroll
    for (int o = 16; o > 0; o >>= 1) p += __shfl_down(p, o, 32);
    if (lane == 0) red[wave] = p;
    __syncthreads();
    if (tid == 0) {
        float s = bv1[0];
        for (int i = 0; i < 8; ++i) s += red[i];
        s_c = s;
    }
    __syncthreads();

    for (int t = wave; t < TT; t += 8) {          // 25 rows per wave, uniform
        float s = 0.0f;
#pragma unroll
        for (int q = 0; q < 8; ++q) s += hb[t * HH + lane + 32 * q] * w1[lane + 32 * q];
#pragma unroll
        for (int o = 16; o > 0; o >>= 1) s += __shfl_down(s, o, 32);
        if (lane == 0) sc[t] = s + s_c;
    }
    __syncthreads();

    float mx = -1e30f;
    for (int t = tid; t < TT; t += 256) mx = fmaxf(mx, sc[t]);
#pragma unroll
    for (int o = 16; o > 0; o >>= 1) mx = fmaxf(mx, __shfl_down(mx, o, 32));
    if (lane == 0) red[wave] = mx;
    __syncthreads();
    if (tid == 0) {
        float m = red[0];
        for (int i = 1; i < 8; ++i) m = fmaxf(m, red[i]);
        s_max = m;
    }
    __syncthreads();

    float ps = 0.0f;
    for (int t = tid; t < TT; t += 256) { float e = fast_exp(sc[t] - s_max); sc[t] = e; ps += e; }
#pragma unroll
    for (int o = 16; o > 0; o >>= 1) ps += __shfl_down(ps, o, 32);
    if (lane == 0) red[wave] = ps;
    __syncthreads();
    if (tid == 0) {
        float s = 0.0f;
        for (int i = 0; i < 8; ++i) s += red[i];
        s_sum = s;
    }
    __syncthreads();

    float accv = 0.0f;
    for (int t = 0; t < TT; ++t) accv += sc[t] * hb[t * HH + tid];
    out[(size_t)b * HH + tid] = accv * __builtin_amdgcn_rcpf(s_sum);
}

// ---------------------------------------------------------------------------
extern "C" void kernel_launch(void* const* d_in, const int* in_sizes, int n_in,
                              void* d_out, int out_size, void* d_ws, size_t ws_size,
                              hipStream_t stream)
{
    const float* inputs = (const float*)d_in[0];
    const float* Wxf = (const float*)d_in[1];
    const float* Wxi = (const float*)d_in[2];
    const float* Wxo = (const float*)d_in[3];
    const float* Wxg = (const float*)d_in[4];
    const float* bxf = (const float*)d_in[5];
    const float* bxi = (const float*)d_in[6];
    const float* bxo = (const float*)d_in[7];
    const float* bxg = (const float*)d_in[8];
    const float* Whf = (const float*)d_in[9];
    const float* Whi = (const float*)d_in[10];
    const float* Who = (const float*)d_in[11];
    const float* Whg = (const float*)d_in[12];
    const float* Wcd = (const float*)d_in[13];
    const float* bcd = (const float*)d_in[14];
    const float* Wv1 = (const float*)d_in[15];
    const float* bv1 = (const float*)d_in[16];

    char* ws = (char*)d_ws;
    __bf16* wcat = (__bf16*)(ws);                    // 384*1024 bf16 = 768 KB
    __bf16* wcdb = (__bf16*)(ws + (768 << 10));      // 256*256  bf16 = 128 KB
    float*  bx   = (float*) (ws + (896 << 10));      // 1024 f32 = 4 KB
    float*  hs   = (float*) (ws + (1 << 20));        // B*T*H f32 = 100 MB

    const int prep_n = KCAT * NGATE + HH * HH + NGATE;
    prep_kernel<<<(prep_n + 255) / 256, 256, 0, stream>>>(
        Wxf, Wxi, Wxo, Wxg, Whf, Whi, Who, Whg, Wcd,
        bxf, bxi, bxo, bxg, wcat, wcdb, bx);

    scan_kernel<<<BB / 16, 256, 0, stream>>>(inputs, wcat, wcdb, bx, bcd, hs);

    pool_kernel<<<BB, 256, 0, stream>>>(hs, Wv1, bv1, (float*)d_out);
}
